// SA_2585570312962
// MI455X (gfx1250) — matrix operand order repaired and checked by reading, not compiled, not run on hardware
//
#include <hip/hip_runtime.h>

// ---------------- CDNA5 WMMA types ----------------
typedef __attribute__((ext_vector_type(16))) __bf16 v16bf;
typedef __attribute__((ext_vector_type(8)))  float  v8f;

// Problem constants (B, C, D, H, W) = (2, 64, 12, 256, 64)
#define NB 2
#define NC 64
#define ND 12
#define NH 256
#define NW 64

#define THREADS 512          // 16 waves of 32
#define NWAVES  16

// LDS layout in bf16 (u16) units. Total = 147456 u16 = 294912 bytes (288 KB <= 320 KB/WGP)
#define OFF_WQ   0
#define OFF_WK   4096
#define OFF_WV   8192
#define OFF_WFC  12288
#define OFF_Q    16384
#define OFF_K    32768
#define OFF_VT   49152
#define OFF_P    65536     /* + wave*4096, 16 waves -> 65536 u16 total */
#define OFF_XO   131072    /* X (bf16) before barrier2; reused as O after */
#define LDS_U16  147456
#define LDS_BYTES (LDS_U16 * 2)

__device__ __forceinline__ unsigned short f32_to_bf16(float f) {
  union { float f; unsigned u; } x; x.f = f;
  unsigned r = x.u + 0x7FFFu + ((x.u >> 16) & 1u);   // round-to-nearest-even
  return (unsigned short)(r >> 16);
}

__device__ __forceinline__ v8f wmma_bf16(v16bf a, v16bf b, v8f c) {
  // D = A(16x32) * B(32x16) + C, f32 accumulate
  return __builtin_amdgcn_wmma_f32_16x16x32_bf16(false, a, false, b, (short)0, c,
                                                 false, false);
}

// A-fragment (16x32 bf16, MxK) from row-major LDS matrix mat[*][ld].
// ISA 7.12.2: lane l -> M = m0 + (l&15); half = l>>4.
//   VGPR j<4 :  K = 2j + 8*half ;  VGPR j>=4 : K = 16 + 2(j-4) + 8*half  (packed pairs)
__device__ __forceinline__ v16bf load_frag_a(const unsigned short* mat, int ld,
                                             int m0, int k0, int lane) {
  const int m = m0 + (lane & 15);
  const int half = lane >> 4;
  const unsigned short* row = mat + (size_t)m * ld + k0;
  union { v16bf v; unsigned u[8]; } f;
#pragma unroll
  for (int j = 0; j < 4; ++j) {
    f.u[j]     = *(const unsigned*)(row + 2 * j + 8 * half);
    f.u[j + 4] = *(const unsigned*)(row + 16 + 2 * j + 8 * half);
  }
  return f.v;
}

// B-fragment (32x16 bf16, KxN) where memory holds B^T row-major: matT[n][k].
// lane l -> N = n0 + (l&15); half = l>>4; VGPR j<4 : K = 2j + 8*half ; VGPR j>=4 : K = 16 + 2(j-4) + 8*half (packed pairs)
__device__ __forceinline__ v16bf load_frag_b(const unsigned short* matT, int ld,
                                             int n0, int k0, int lane) {
  const int n = n0 + (lane & 15);
  const int half = lane >> 4;
  const unsigned short* row = matT + (size_t)n * ld + k0;
  union { v16bf v; unsigned u[8]; } f;
#pragma unroll
  for (int j = 0; j < 4; ++j) {
    f.u[j]     = *(const unsigned*)(row + 2 * j + 8 * half);
    f.u[j + 4] = *(const unsigned*)(row + 16 + 2 * j + 8 * half);
  }
  return f.v;
}

// C/D layout (ISA 7.12.2): VGPR r, lane l -> row = r + 8*(l>>4), col = l&15.
__device__ __forceinline__ void store_tile_relu_bias(unsigned short* dst, int ld,
                                                     int m0, int n0, const v8f c,
                                                     const float* bias, int lane) {
  const int col = lane & 15, half = lane >> 4;
  const float bv = bias[n0 + col];
#pragma unroll
  for (int r = 0; r < 8; ++r) {
    float v = c[r] + bv;
    v = v > 0.f ? v : 0.f;
    dst[(size_t)(m0 + r + 8 * half) * ld + n0 + col] = f32_to_bf16(v);
  }
}

// Same, but store transposed (for V^T staging): dst[n][m], ld = row length of dst.
__device__ __forceinline__ void store_tile_relu_bias_T(unsigned short* dst, int ld,
                                                       int m0, int n0, const v8f c,
                                                       const float* bias, int lane) {
  const int col = lane & 15, half = lane >> 4;
  const float bv = bias[n0 + col];
#pragma unroll
  for (int r = 0; r < 8; ++r) {
    float v = c[r] + bv;
    v = v > 0.f ? v : 0.f;
    dst[(size_t)(n0 + col) * ld + (m0 + r + 8 * half)] = f32_to_bf16(v);
  }
}

__device__ __forceinline__ void store_tile_plain(unsigned short* dst, int ld,
                                                 int m0, int n0, const v8f c, int lane) {
  const int col = lane & 15, half = lane >> 4;
#pragma unroll
  for (int r = 0; r < 8; ++r)
    dst[(size_t)(m0 + r + 8 * half) * ld + n0 + col] = f32_to_bf16(c[r]);
}

__global__ __launch_bounds__(THREADS, 1)
void SA_fused_attn_kernel(const float* __restrict__ rep,
                          const float* __restrict__ wq, const float* __restrict__ bq,
                          const float* __restrict__ wk, const float* __restrict__ bk,
                          const float* __restrict__ wv, const float* __restrict__ bv,
                          const float* __restrict__ wfc, const float* __restrict__ bfc,
                          float* __restrict__ out) {
  extern __shared__ unsigned short lds[];
  unsigned short* WQl  = lds + OFF_WQ;
  unsigned short* WKl  = lds + OFF_WK;
  unsigned short* WVl  = lds + OFF_WV;
  unsigned short* WFCl = lds + OFF_WFC;
  unsigned short* Ql   = lds + OFF_Q;    // 256 x 64 (h x c)
  unsigned short* Kl   = lds + OFF_K;    // 256 x 64 (g x c)
  unsigned short* VTl  = lds + OFF_VT;   // 64 x 256 (c x g)   == V^T
  unsigned short* XOl  = lds + OFF_XO;   // 256 x 64: X^T then O

  const int tid  = threadIdx.x;
  const int lane = tid & 31;
  const int wave = tid >> 5;
  const int ho   = wave * 16;            // this wave's H-row tile base
  const int col  = lane & 15;
  const int half = lane >> 4;

  // slice decode: grid = B*D*W
  const int s = blockIdx.x;
  const int w = s & (NW - 1);
  const int d = (s >> 6) % ND;
  const int b = s / (NW * ND);

  // rep index: ((b*C + c)*D + d)*H*W + h*W + w
  const size_t CSTR = (size_t)ND * NH * NW;        // 196608
  const size_t base = (size_t)b * NC * CSTR + (size_t)d * NH * NW + w;

  // ---- stage weights (f32 -> bf16) ----
#pragma unroll 2
  for (int i = tid; i < NC * NC; i += THREADS) {
    WQl[i]  = f32_to_bf16(wq[i]);
    WKl[i]  = f32_to_bf16(wk[i]);
    WVl[i]  = f32_to_bf16(wv[i]);
    WFCl[i] = f32_to_bf16(wfc[i]);
  }
  // ---- stage X^T (h x c): Xt[h][c] = rep[b,c,d,h,w] ----
  for (int i = tid; i < NC * NH; i += THREADS) {
    const int h = i & (NH - 1);
    const int c = i >> 8;
    XOl[(size_t)h * NC + c] = f32_to_bf16(rep[base + (size_t)c * CSTR + (size_t)h * NW]);
  }
  __syncthreads();

  // ================= GEMM1: Q/K/V projections (wave-local 16-row tile) =========
  {
    const v16bf xa0 = load_frag_a(XOl, NC, ho, 0,  lane);
    const v16bf xa1 = load_frag_a(XOl, NC, ho, 32, lane);
#pragma unroll
    for (int nt = 0; nt < 4; ++nt) {
      v8f cq = {}, ck = {}, cv = {};
      cq = wmma_bf16(xa0, load_frag_b(WQl, NC, nt * 16, 0,  lane), cq);
      cq = wmma_bf16(xa1, load_frag_b(WQl, NC, nt * 16, 32, lane), cq);
      ck = wmma_bf16(xa0, load_frag_b(WKl, NC, nt * 16, 0,  lane), ck);
      ck = wmma_bf16(xa1, load_frag_b(WKl, NC, nt * 16, 32, lane), ck);
      cv = wmma_bf16(xa0, load_frag_b(WVl, NC, nt * 16, 0,  lane), cv);
      cv = wmma_bf16(xa1, load_frag_b(WVl, NC, nt * 16, 32, lane), cv);
      store_tile_relu_bias  (Ql,  NC, ho, nt * 16, cq, bq, lane);
      store_tile_relu_bias  (Kl,  NC, ho, nt * 16, ck, bk, lane);
      store_tile_relu_bias_T(VTl, NH, ho, nt * 16, cv, bv, lane);
    }
  }
  __syncthreads();   // K, V^T now visible to all waves; X no longer needed

  // ================= GEMM2 + streaming softmax =================================
  // Two-pass over S = Q K^T (recompute instead of holding 16 v8f accumulators:
  // 32 extra WMMAs per wave is free vs. the spill traffic they'd otherwise cost).
  unsigned short* Pl = lds + OFF_P + wave * (16 * NH);  // wave-private 16 x 256
  float inv[8];
  {
    const v16bf qa0 = load_frag_a(Ql, NC, ho, 0,  lane);
    const v16bf qa1 = load_frag_a(Ql, NC, ho, 32, lane);

    // pass 1: per-row max of raw scores
    float mx[8];
#pragma unroll
    for (int r = 0; r < 8; ++r) mx[r] = -3.402823e38f;
#pragma unroll
    for (int nt = 0; nt < 16; ++nt) {
      v8f c = {};
      c = wmma_bf16(qa0, load_frag_b(Kl, NC, nt * 16, 0,  lane), c);
      c = wmma_bf16(qa1, load_frag_b(Kl, NC, nt * 16, 32, lane), c);
#pragma unroll
      for (int r = 0; r < 8; ++r) mx[r] = fmaxf(mx[r], c[r]);
    }
#pragma unroll
    for (int r = 0; r < 8; ++r)
#pragma unroll
      for (int off = 1; off < 16; off <<= 1)
        mx[r] = fmaxf(mx[r], __shfl_xor(mx[r], off, 32));

    // pass 2: recompute, exp, accumulate row sums, store UNNORMALIZED P (bf16).
    // softmax(s/8) = exp((s-mx)*0.125)/sum; the 1/sum is applied to O rows later.
    float sum[8];
#pragma unroll
    for (int r = 0; r < 8; ++r) sum[r] = 0.f;
#pragma unroll
    for (int nt = 0; nt < 16; ++nt) {
      v8f c = {};
      c = wmma_bf16(qa0, load_frag_b(Kl, NC, nt * 16, 0,  lane), c);
      c = wmma_bf16(qa1, load_frag_b(Kl, NC, nt * 16, 32, lane), c);
#pragma unroll
      for (int r = 0; r < 8; ++r) {
        const float p = __expf((c[r] - mx[r]) * 0.125f);
        sum[r] += p;
        Pl[(size_t)(r + 8 * half) * NH + nt * 16 + col] = f32_to_bf16(p);
      }
    }
#pragma unroll
    for (int r = 0; r < 8; ++r) {
#pragma unroll
      for (int off = 1; off < 16; off <<= 1) sum[r] += __shfl_xor(sum[r], off, 32);
      inv[r] = __frcp_rn(sum[r]);
    }
  }

  // ================= GEMM3: O = (P V) * inv_row  (16 x 64), K-dim = 256 ========
  {
    v8f oc0 = {}, oc1 = {}, oc2 = {}, oc3 = {};
#pragma unroll
    for (int ks = 0; ks < 8; ++ks) {
      const v16bf pa = load_frag_a(Pl, NH, 0, ks * 32, lane);
      oc0 = wmma_bf16(pa, load_frag_b(VTl, NH, 0,  ks * 32, lane), oc0);
      oc1 = wmma_bf16(pa, load_frag_b(VTl, NH, 16, ks * 32, lane), oc1);
      oc2 = wmma_bf16(pa, load_frag_b(VTl, NH, 32, ks * 32, lane), oc2);
      oc3 = wmma_bf16(pa, load_frag_b(VTl, NH, 48, ks * 32, lane), oc3);
    }
    // apply deferred softmax normalization per output row (row = r + 8*half)
#pragma unroll
    for (int r = 0; r < 8; ++r) {
      oc0[r] *= inv[r]; oc1[r] *= inv[r]; oc2[r] *= inv[r]; oc3[r] *= inv[r];
    }
    store_tile_plain(XOl, NC, ho, 0,  oc0, lane);   // X slab reused as O
    store_tile_plain(XOl, NC, ho, 16, oc1, lane);
    store_tile_plain(XOl, NC, ho, 32, oc2, lane);
    store_tile_plain(XOl, NC, ho, 48, oc3, lane);
  }

  // ================= GEMM4: Y = relu(O FC^T + b), write (B,C,D,H,W) ============
  {
    const v16bf oa0 = load_frag_a(XOl, NC, ho, 0,  lane);
    const v16bf oa1 = load_frag_a(XOl, NC, ho, 32, lane);
#pragma unroll
    for (int nt = 0; nt < 4; ++nt) {
      v8f c = {};
      c = wmma_bf16(oa0, load_frag_b(WFCl, NC, nt * 16, 0,  lane), c);
      c = wmma_bf16(oa1, load_frag_b(WFCl, NC, nt * 16, 32, lane), c);
      const int o = nt * 16 + col;
      const float bv = bfc[o];
      const size_t obase = base + (size_t)o * CSTR;
#pragma unroll
      for (int r = 0; r < 8; ++r) {
        const int h = ho + r + 8 * half;
        float v = c[r] + bv;
        v = v > 0.f ? v : 0.f;
        out[obase + (size_t)h * NW] = v;
      }
    }
  }
}

extern "C" void kernel_launch(void* const* d_in, const int* in_sizes, int n_in,
                              void* d_out, int out_size, void* d_ws, size_t ws_size,
                              hipStream_t stream) {
  const float* rep = (const float*)d_in[0];
  const float* wq  = (const float*)d_in[1];
  const float* bq  = (const float*)d_in[2];
  const float* wk  = (const float*)d_in[3];
  const float* bk  = (const float*)d_in[4];
  const float* wv  = (const float*)d_in[5];
  const float* bv  = (const float*)d_in[6];
  const float* wfc = (const float*)d_in[7];
  const float* bfc = (const float*)d_in[8];
  float* out = (float*)d_out;

  (void)in_sizes; (void)n_in; (void)out_size; (void)d_ws; (void)ws_size;

  static_assert(LDS_BYTES == 294912, "LDS budget");
  hipFuncSetAttribute((const void*)SA_fused_attn_kernel,
                      hipFuncAttributeMaxDynamicSharedMemorySize, LDS_BYTES);

  const int blocks = NB * ND * NW;   // 1536 independent (b,d,w) slices
  SA_fused_attn_kernel<<<blocks, THREADS, LDS_BYTES, stream>>>(
      rep, wq, bq, wk, bk, wv, bv, wfc, bfc, out);
}
